// HybridSelfAttentionTransformer_65481071396697
// MI455X (gfx1250) — matrix operand order repaired and checked by reading, not compiled, not run on hardware
//
#include <hip/hip_runtime.h>
#include <hip/hip_bf16.h>
#include <math.h>

// ---------------------------------------------------------------------------
// HybridSelfAttentionTransformer forward, MI455X (gfx1250, wave32, WMMA)
// B=8 S=1024 D=512 H=8 dk=64 F=2048 L=2 C=10
// All matmuls in bf16 WMMA (v_wmma_f32_16x16x32_bf16), fp32 accumulate.
// Dense GEMMs use double-buffered LDS staging via gfx1250 async
// global->LDS DMA (GLOBAL_LOAD_ASYNC_TO_LDS_B128 + s_wait_asynccnt) when the
// toolchain exposes the builtins; otherwise falls back to sync staging.
// ---------------------------------------------------------------------------

typedef __attribute__((ext_vector_type(16))) __bf16 v16bf;
typedef __attribute__((ext_vector_type(8)))  float  v8f;

#define B_  8
#define S_  1024
#define D_  512
#define H_  8
#define DK_ 64
#define F_  2048
#define L_  2
#define C_  10
#define BS_ (B_ * S_)

#if __has_builtin(__builtin_amdgcn_global_load_async_to_lds_b128) && \
    __has_builtin(__builtin_amdgcn_s_wait_asynccnt)
#define HAVE_ASYNC 1
#else
#define HAVE_ASYNC 0
#endif

#if HAVE_ASYNC
// Builtin signature (from hipcc diagnostic): param1 = AS1 (global) pointer to
// __attribute__((vector_size(16))) int, param2 = AS3 (LDS) analogue.
typedef int v4i_ __attribute__((vector_size(16)));
typedef __attribute__((address_space(1))) v4i_ gv4i_t;   // global int4
typedef __attribute__((address_space(3))) v4i_ lv4i_t;   // LDS int4 (32-bit ptr)
#endif

// Stage 16 bytes global -> LDS (async DMA when available).
__device__ __forceinline__ void stage_b128(const void* g, void* l) {
#if HAVE_ASYNC
  // low 32 bits of a generic pointer into __shared__ are the LDS offset
  __builtin_amdgcn_global_load_async_to_lds_b128(
      (gv4i_t*)(unsigned long long)g,
      (lv4i_t*)(unsigned)(unsigned long long)l, 0, 0);
#else
  *(uint4*)l = *(const uint4*)g;
#endif
}

template <int N>
__device__ __forceinline__ void wait_stage() {
#if HAVE_ASYNC
  __builtin_amdgcn_s_wait_asynccnt(N);
#endif
}

union FragB16 { v16bf v; unsigned u[8]; unsigned short s16[16]; __bf16 h[16]; };

__device__ __forceinline__ v8f zero8() {
  v8f z = {0.f, 0.f, 0.f, 0.f, 0.f, 0.f, 0.f, 0.f};
  return z;
}

// A-matrix (16x32, MxK) per-lane K mapping (ISA 7.12.2, 16-bit A):
// lanes 0-15: V0..3 -> K=0..7, V4..7 -> K=16..23 ; lanes 16-31: +8
__device__ __forceinline__ int ka_of(int p, int half) {
  return (p < 4) ? (8 * half + 2 * p) : (16 + 8 * half + 2 * (p - 4));
}
// B-matrix (32x16, KxN): lane = N, same K mapping as A (lanes 16-31: +8)
__device__ __forceinline__ int kb_of(int p, int half) {
  return ka_of(p, half);
}

// A fragment from row-major bf16 source: element (m,k) at base[m*ld + k]
__device__ __forceinline__ v16bf load_frag_a(const __bf16* base, int ld) {
  const int lane = threadIdx.x & 31, half = lane >> 4, m = lane & 15;
  FragB16 f;
  const __bf16* rp = base + m * ld;
#pragma unroll
  for (int p = 0; p < 8; ++p) {
    f.u[p] = *(const unsigned*)(rp + ka_of(p, half));  // K pairs contiguous
  }
  return f.v;
}

// B fragment where the source holds B transposed (element (k,n) at
// base[n*ld + k]): k contiguous per lane -> dword loads only.
__device__ __forceinline__ v16bf load_frag_bt(const __bf16* base, int ld) {
  const int lane = threadIdx.x & 31, half = lane >> 4, n = lane & 15;
  FragB16 f;
  const __bf16* rp = base + n * ld;
#pragma unroll
  for (int p = 0; p < 8; ++p) {
    f.u[p] = *(const unsigned*)(rp + kb_of(p, half));
  }
  return f.v;
}

// A fragment built from fp32 LDS scores, scaled per-row by invp[m], cvt->bf16
__device__ __forceinline__ v16bf load_frag_p(const float* sc, int ld, const float* invp) {
  const int lane = threadIdx.x & 31, half = lane >> 4, m = lane & 15;
  const float inv = invp[m];
  FragB16 f;
#pragma unroll
  for (int p = 0; p < 8; ++p) {
    int k = ka_of(p, half);
    f.h[2 * p]     = (__bf16)(sc[m * ld + k] * inv);
    f.h[2 * p + 1] = (__bf16)(sc[m * ld + k + 1] * inv);
  }
  return f.v;
}

__device__ __forceinline__ v8f wmma_bf16(v16bf a, v16bf b, v8f c) {
  return __builtin_amdgcn_wmma_f32_16x16x32_bf16(false, a, false, b, (short)0, c,
                                                 false, false);
}

// ---------------------------------------------------------------------------
// Elementwise / data-layout kernels
// ---------------------------------------------------------------------------

// out[n*K + k] = in[k*N + n]  (fp32 -> bf16, transpose KxN -> NxK)
__global__ void k_cvt_t(const float* __restrict__ in, __bf16* __restrict__ out,
                        int K, int N) {
  int i = blockIdx.x * blockDim.x + threadIdx.x;
  if (i >= K * N) return;
  int n = i / K, k = i - n * K;
  out[i] = (__bf16)in[(size_t)k * N + n];
}

// x[r, d] = embedding[tok[r], d] + PE(r % S, d)
__global__ void k_embed(const int* __restrict__ tokens,
                        const float* __restrict__ emb,
                        float* __restrict__ x) {
  const int row = blockIdx.x;
  const int s = row & (S_ - 1);
  const int tok = tokens[row];
  const float* er = emb + (size_t)tok * D_;
  float* xr = x + (size_t)row * D_;
#pragma unroll
  for (int i = 0; i < 2; ++i) {
    int d = threadIdx.x + i * 256;
    int j = d >> 1;
    float ang = (float)s * __expf(-((float)(2 * j) / (float)D_) * 9.210340371976184f);
    float pe = (d & 1) ? __cosf(ang) : __sinf(ang);
    xr[d] = er[d] + pe;
  }
}

// p[r, d] = bf16(cos(x[r,d] + theta[d]))
__global__ void k_cosproj(const float* __restrict__ x,
                          const float* __restrict__ theta,
                          __bf16* __restrict__ p) {
  const int row = blockIdx.x;
  const float* xr = x + (size_t)row * D_;
  __bf16* pr = p + (size_t)row * D_;
#pragma unroll
  for (int i = 0; i < 2; ++i) {
    int d = threadIdx.x + i * 256;
    pr[d] = (__bf16)__cosf(xr[d] + theta[d]);
  }
}

// pT[((b*H + h)*DK + d)*S + s] = p[(b*S + s)*D + h*DK + d]
__global__ void k_transpose_p(const __bf16* __restrict__ p, __bf16* __restrict__ pT) {
  const int s = blockIdx.x * 256 + threadIdx.x;
  const int d = blockIdx.y;        // 0..DK-1
  const int bh = blockIdx.z;       // 0..B*H-1
  const int b = bh >> 3, h = bh & 7;
  pT[((size_t)bh * DK_ + d) * S_ + s] =
      p[((size_t)b * S_ + s) * D_ + h * DK_ + d];
}

// x[r,:] = LN(x[r,:] + res[r,:]) * g + b   (one wave per row, wave32)
__global__ void k_residual_ln(float* __restrict__ x,
                              const float* __restrict__ res,
                              const float* __restrict__ g,
                              const float* __restrict__ b) {
  const int wave = threadIdx.x >> 5;
  const int lane = threadIdx.x & 31;
  const int row = blockIdx.x * 8 + wave;
  float* xr = x + (size_t)row * D_;
  const float* rr = res + (size_t)row * D_;
  float v[16];
  float sum = 0.f;
#pragma unroll
  for (int i = 0; i < 16; ++i) {
    int c = lane + i * 32;
    v[i] = xr[c] + rr[c];
    sum += v[i];
  }
#pragma unroll
  for (int o = 16; o; o >>= 1) sum += __shfl_xor(sum, o);
  const float mu = sum * (1.f / (float)D_);
  float s2 = 0.f;
#pragma unroll
  for (int i = 0; i < 16; ++i) { float t = v[i] - mu; s2 += t * t; }
#pragma unroll
  for (int o = 16; o; o >>= 1) s2 += __shfl_xor(s2, o);
  const float rstd = rsqrtf(s2 * (1.f / (float)D_) + 1e-5f);
#pragma unroll
  for (int i = 0; i < 16; ++i) {
    int c = lane + i * 32;
    xr[c] = g[c] * (v[i] - mu) * rstd + b[c];
  }
}

__global__ void k_pool(const float* __restrict__ x, float* __restrict__ pooled) {
  int idx = blockIdx.x * blockDim.x + threadIdx.x;   // [0, B*D)
  if (idx >= B_ * D_) return;
  int b = idx / D_, d = idx - b * D_;
  const float* xp = x + ((size_t)b * S_) * D_ + d;
  float sum = 0.f;
  for (int s = 0; s < S_; ++s) sum += xp[(size_t)s * D_];
  pooled[idx] = sum * (1.f / (float)S_);
}

__global__ void k_cls(const float* __restrict__ pooled,
                      const float* __restrict__ Wcls,
                      const float* __restrict__ bcls,
                      float* __restrict__ out) {
  int t = threadIdx.x;
  if (t >= B_ * C_) return;
  int b = t / C_, c = t - b * C_;
  float acc = bcls[c];
  const float* pr = pooled + (size_t)b * D_;
  for (int d = 0; d < D_; ++d) acc += pr[d] * Wcls[(size_t)d * C_ + c];
  out[t] = acc;
}

// ---------------------------------------------------------------------------
// Fused attention: q == k == v == p (shared quantum layer).
// Block = 64 threads (2 waves); each wave owns 16 query rows of one (b,h).
// Scores row-block (16 x 1024, fp32) lives in dynamic LDS per wave.
// PV value fragments come from the pre-transposed pT (contiguous dwords).
// ---------------------------------------------------------------------------
__global__ void k_attn(const __bf16* __restrict__ p, const __bf16* __restrict__ pT,
                       __bf16* __restrict__ out) {
  extern __shared__ float smem[];
  const int wave = threadIdx.x >> 5;
  const int lane = threadIdx.x & 31;
  const int half = lane >> 4;
  float* sc = smem + (size_t)wave * 16 * S_;
  float* invp = smem + 2 * 16 * S_ + wave * 16;

  const int qbase = (blockIdx.x * 2 + wave) * 16;
  const int h = blockIdx.y, b = blockIdx.z;
  const __bf16* ph = p + ((size_t)b * S_) * D_ + h * DK_;      // (s, d)
  const __bf16* phT = pT + ((size_t)(b * H_ + h) * DK_) * S_;  // (d, s)

  // Q fragments for the two K=32 chunks of dk=64
  v16bf q0 = load_frag_a(ph + (size_t)qbase * D_ + 0, D_);
  v16bf q1 = load_frag_a(ph + (size_t)qbase * D_ + 32, D_);

  // scores = (Q Kt) / sqrt(dk); K-tile acts as B = (key rows)^T
  for (int nt = 0; nt < S_ / 16; ++nt) {
    const __bf16* kb = ph + (size_t)(nt * 16) * D_;
    v16bf b0 = load_frag_bt(kb + 0, D_);
    v16bf b1 = load_frag_bt(kb + 32, D_);
    v8f acc = zero8();
    acc = wmma_bf16(q0, b0, acc);
    acc = wmma_bf16(q1, b1, acc);
#pragma unroll
    for (int r = 0; r < 8; ++r) {
      int m = r + 8 * half;
      sc[m * S_ + nt * 16 + (lane & 15)] = acc[r] * 0.125f;  // 1/sqrt(64)
    }
  }

  // in-wave softmax over each of the 16 rows (keep exp unnormalized, save 1/sum)
  for (int r = 0; r < 16; ++r) {
    float mx = -3.4e38f;
    for (int c = lane; c < S_; c += 32) mx = fmaxf(mx, sc[r * S_ + c]);
#pragma unroll
    for (int o = 16; o; o >>= 1) mx = fmaxf(mx, __shfl_xor(mx, o));
    float sum = 0.f;
    for (int c = lane; c < S_; c += 32) {
      float e = __expf(sc[r * S_ + c] - mx);
      sc[r * S_ + c] = e;
      sum += e;
    }
#pragma unroll
    for (int o = 16; o; o >>= 1) sum += __shfl_xor(sum, o);
    if (lane == 0) invp[r] = 1.f / sum;
  }

  // O = P V : A = probs (16 x 1024 over keys), B[k][n] = pT[(d0+n)*S + k]
#pragma unroll
  for (int dt = 0; dt < 4; ++dt) {
    v8f acc = zero8();
    for (int kt = 0; kt < S_ / 32; ++kt) {
      v16bf a = load_frag_p(sc + kt * 32, S_, invp);
      v16bf bv = load_frag_bt(phT + (size_t)(dt * 16) * S_ + kt * 32, S_);
      acc = wmma_bf16(a, bv, acc);
    }
#pragma unroll
    for (int r = 0; r < 8; ++r) {
      int m = r + 8 * half;
      size_t row = (size_t)b * S_ + qbase + m;
      out[row * D_ + h * DK_ + dt * 16 + (lane & 15)] = (__bf16)acc[r];
    }
  }
}

// ---------------------------------------------------------------------------
// Dense GEMM: C[M,N] = A[M,K](bf16) * W[K,N] + bias, with W supplied
// TRANSPOSED (Wt[n*K + k]).  Block 128 threads (4 waves), tile 128x64,
// K-step 32, double-buffered async LDS staging; each wave owns a 32x64 slab
// (8 WMMAs per K-step).  Optional ReLU; fp32 or bf16 output.
// ---------------------------------------------------------------------------
template <bool RELU, bool OUTBF16>
__global__ void k_gemm(const __bf16* __restrict__ A, const __bf16* __restrict__ Wt,
                       const float* __restrict__ bias,
                       float* __restrict__ Cf, __bf16* __restrict__ Cb,
                       int M, int N, int K) {
  __shared__ __align__(16) __bf16 As[2][128][32];   // 16 KB
  __shared__ __align__(16) __bf16 Bst[2][64][32];   //  8 KB
  const int t = threadIdx.x;
  const int wave = t >> 5;
  const int lane = t & 31;
  const int half = lane >> 4;
  const int m0 = blockIdx.y * 128;
  const int n0 = blockIdx.x * 64;

  // per-thread staging coordinates (6 b128 transfers per thread per slab)
  const int ar0 = t >> 2, ac0 = (t & 3) * 8;              // As rows t/4, +32, +64, +96
  const int br0 = t >> 2, bc0 = (t & 3) * 8;              // Bst rows t/4, +32

  auto stage_slab = [&](int k0, int buf) {
#pragma unroll
    for (int i = 0; i < 4; ++i) {
      int ar = ar0 + i * 32;
      stage_b128(A + (size_t)(m0 + ar) * K + k0 + ac0, &As[buf][ar][ac0]);
    }
#pragma unroll
    for (int i = 0; i < 2; ++i) {
      int br = br0 + i * 32;
      stage_b128(Wt + (size_t)(n0 + br) * K + k0 + bc0, &Bst[buf][br][bc0]);
    }
  };

  v8f acc[2][4];
#pragma unroll
  for (int i = 0; i < 2; ++i)
#pragma unroll
    for (int j = 0; j < 4; ++j) acc[i][j] = zero8();

  const int nslab = K >> 5;
  stage_slab(0, 0);
  for (int s = 0; s < nslab; ++s) {
    const int cur = s & 1;
    if (s + 1 < nslab) {
      stage_slab((s + 1) << 5, cur ^ 1);
      wait_stage<6>();   // slab s complete; slab s+1 may remain in flight
    } else {
      wait_stage<0>();
    }
    __syncthreads();

    v16bf af0 = load_frag_a(&As[cur][wave * 32][0], 32);
    v16bf af1 = load_frag_a(&As[cur][wave * 32 + 16][0], 32);
#pragma unroll
    for (int nt = 0; nt < 4; ++nt) {
      v16bf bf = load_frag_bt(&Bst[cur][nt * 16][0], 32);
      acc[0][nt] = wmma_bf16(af0, bf, acc[0][nt]);
      acc[1][nt] = wmma_bf16(af1, bf, acc[1][nt]);
    }
    __syncthreads();   // buf `cur` free for slab s+2 staging next iteration
  }

#pragma unroll
  for (int sub = 0; sub < 2; ++sub) {
#pragma unroll
    for (int nt = 0; nt < 4; ++nt) {
#pragma unroll
      for (int r = 0; r < 8; ++r) {
        int m = m0 + wave * 32 + sub * 16 + r + 8 * half;
        int n = n0 + nt * 16 + (lane & 15);
        float v = acc[sub][nt][r] + bias[n];
        if (RELU) v = fmaxf(v, 0.f);
        if (OUTBF16) Cb[(size_t)m * N + n] = (__bf16)v;
        else         Cf[(size_t)m * N + n] = v;
      }
    }
  }
}

// ---------------------------------------------------------------------------
// Host-side orchestration
// ---------------------------------------------------------------------------
extern "C" void kernel_launch(void* const* d_in, const int* in_sizes, int n_in,
                              void* d_out, int out_size, void* d_ws, size_t ws_size,
                              hipStream_t stream) {
  (void)in_sizes; (void)n_in; (void)out_size; (void)ws_size;
  const int*   tokens     = (const int*)d_in[0];
  const float* embedding  = (const float*)d_in[1];
  const float* theta_attn = (const float*)d_in[2];
  const float* Wc         = (const float*)d_in[3];
  const float* bc         = (const float*)d_in[4];
  const float* ln1_g      = (const float*)d_in[5];
  const float* ln1_b      = (const float*)d_in[6];
  const float* theta_ffn  = (const float*)d_in[7];
  const float* W1         = (const float*)d_in[8];
  const float* b1         = (const float*)d_in[9];
  const float* W2         = (const float*)d_in[10];
  const float* b2         = (const float*)d_in[11];
  const float* ln2_g      = (const float*)d_in[12];
  const float* ln2_b      = (const float*)d_in[13];
  const float* Wcls       = (const float*)d_in[14];
  const float* bcls       = (const float*)d_in[15];
  // d_in[16] = num_heads (hardcoded 8)

  char* ws = (char*)d_ws;
  const size_t MB = (size_t)1 << 20;
  float*  x      = (float*)(ws);                 // 16 MB  x activations
  float*  tmp    = (float*)(ws + 16 * MB);       // 16 MB  GEMM fp32 out
  __bf16* pbuf   = (__bf16*)(ws + 32 * MB);      //  8 MB  cos-projection bf16
  __bf16* abuf   = (__bf16*)(ws + 40 * MB);      //  8 MB  attn output bf16
  __bf16* hbuf   = (__bf16*)(ws + 48 * MB);      // 32 MB  FFN hidden bf16
  __bf16* pT     = (__bf16*)(ws + 80 * MB);      //  8 MB  transposed values
  __bf16* Wc_t   = (__bf16*)(ws + 88 * MB);      //  1 MB  [N=D][K=D] x L
  __bf16* W1_t   = (__bf16*)(ws + 89 * MB);      //  4 MB  [N=F][K=D] x L
  __bf16* W2_t   = (__bf16*)(ws + 93 * MB);      //  4 MB  [N=D][K=F] x L
  float*  pooled = (float*)(ws + 97 * MB);       // 16 KB

  // fp32 -> bf16 transposed weight copies (per layer)
  for (int l = 0; l < L_; ++l) {
    k_cvt_t<<<(D_ * D_ + 255) / 256, 256, 0, stream>>>(
        Wc + (size_t)l * D_ * D_, Wc_t + (size_t)l * D_ * D_, D_, D_);
    k_cvt_t<<<(D_ * F_ + 255) / 256, 256, 0, stream>>>(
        W1 + (size_t)l * D_ * F_, W1_t + (size_t)l * D_ * F_, D_, F_);
    k_cvt_t<<<(F_ * D_ + 255) / 256, 256, 0, stream>>>(
        W2 + (size_t)l * F_ * D_, W2_t + (size_t)l * F_ * D_, F_, D_);
  }

  k_embed<<<BS_, 256, 0, stream>>>(tokens, embedding, x);

  const size_t attn_smem = (2 * 16 * S_ + 32) * sizeof(float);  // 128 KB + inv
  for (int l = 0; l < L_; ++l) {
    // --- attention block ---
    k_cosproj<<<BS_, 256, 0, stream>>>(x, theta_attn + (size_t)l * D_, pbuf);
    k_transpose_p<<<dim3(S_ / 256, DK_, B_ * H_), 256, 0, stream>>>(pbuf, pT);
    dim3 ag(S_ / 32, H_, B_);
    k_attn<<<ag, 64, attn_smem, stream>>>(pbuf, pT, abuf);
    k_gemm<false, false><<<dim3(D_ / 64, BS_ / 128), 128, 0, stream>>>(
        abuf, Wc_t + (size_t)l * D_ * D_, bc + (size_t)l * D_, tmp, nullptr,
        BS_, D_, D_);
    k_residual_ln<<<BS_ / 8, 256, 0, stream>>>(x, tmp, ln1_g + (size_t)l * D_,
                                               ln1_b + (size_t)l * D_);
    // --- FFN block ---
    k_cosproj<<<BS_, 256, 0, stream>>>(x, theta_ffn + (size_t)l * D_, pbuf);
    k_gemm<true, true><<<dim3(F_ / 64, BS_ / 128), 128, 0, stream>>>(
        pbuf, W1_t + (size_t)l * D_ * F_, b1 + (size_t)l * F_, nullptr, hbuf,
        BS_, F_, D_);
    k_gemm<false, false><<<dim3(D_ / 64, BS_ / 128), 128, 0, stream>>>(
        hbuf, W2_t + (size_t)l * F_ * D_, b2 + (size_t)l * D_, tmp, nullptr,
        BS_, D_, F_);
    k_residual_ln<<<BS_ / 8, 256, 0, stream>>>(x, tmp, ln2_g + (size_t)l * D_,
                                               ln2_b + (size_t)l * D_);
  }

  k_pool<<<(B_ * D_ + 255) / 256, 256, 0, stream>>>(x, pooled);
  k_cls<<<1, 128, 0, stream>>>(pooled, Wcls, bcls, (float*)d_out);
}
